// PatchEncoderNN_46583215292471
// MI455X (gfx1250) — matrix operand order repaired and checked by reading, not compiled, not run on hardware
//
#include <hip/hip_runtime.h>
#include <hip/hip_bf16.h>
#include <math.h>

typedef __attribute__((ext_vector_type(16))) __bf16 v16bf;
typedef __attribute__((ext_vector_type(8)))  float  v8f;

#define H0   128
#define H1   256
#define CNUM 1024
#define EPSL 1e-5f

__device__ __forceinline__ float gelu_exact(float v) {
    return 0.5f * v * (1.0f + erff(v * 0.70710678118654752f));
}

// B-fragment: packed bf16 weights in lane order; lane's 16 elems contiguous (32B aligned)
__device__ __forceinline__ v16bf load_b_frag(const __bf16* wp, int kt, int nt, int NT, int lane) {
    return *(const v16bf*)(wp + ((((kt * NT + nt) * 32) + lane) << 4));
}

// A-fragment from LDS fp32 tile [16][stride], rows at column offset xoff.
// contiguous A k order used here (matches pack_w_bf16): M = lane&15; K = kb + i + 16*(lane>=16)
__device__ __forceinline__ v16bf load_a_frag(const float* sh, int stride, int xoff, int kb, int lane) {
    const int am = lane & 15;
    const int ah = (lane >> 4) << 4;
    v16bf a;
#pragma unroll
    for (int i = 0; i < 16; ++i) {
        int k = kb + i + ah;
        a[i] = (__bf16)sh[am * stride + xoff + k];
    }
    return a;
}

// conv1 MLP for a 16-point tile: x = Lin2(gelu(LN(Lin1(p))))  -> written to sh[:, xoff:xoff+128]
__device__ __forceinline__ void compute_x_tile(
    float* sh, int stride, int xoff, float* stats,
    long long p0, int lane,
    const float* pts, const float* w1, const float* b1,
    const float* g1, const float* beta1,
    const __bf16* w2p, const float* b2)
{
    const int m  = lane & 15;
    const int hf = lane >> 4;
    // Linear(6,128): each lane does 64 outputs for its point
    float pv[6];
#pragma unroll
    for (int k = 0; k < 6; ++k) pv[k] = pts[(p0 + m) * 6 + k];
#pragma unroll 4
    for (int j = 0; j < 64; ++j) {
        int n = hf * 64 + j;
        float acc = b1[n];
#pragma unroll
        for (int k = 0; k < 6; ++k) acc += pv[k] * w1[k * H0 + n];
        sh[m * stride + xoff + n] = acc;
    }
    __syncthreads();
    // LayerNorm(128) stats
    if (lane < 16) {
        float s = 0.f, s2 = 0.f;
        for (int n = 0; n < H0; ++n) {
            float v = sh[lane * stride + xoff + n];
            s += v; s2 += v * v;
        }
        float mu  = s * (1.0f / H0);
        float var = s2 * (1.0f / H0) - mu * mu;
        stats[lane]      = mu;
        stats[16 + lane] = rsqrtf(var + EPSL);
    }
    __syncthreads();
    {   // normalize + affine + exact GELU, in place
        float mu = stats[m], rs = stats[16 + m];
#pragma unroll 4
        for (int j = 0; j < 64; ++j) {
            int n = hf * 64 + j;
            int o = m * stride + xoff + n;
            float v = sh[o];
            v = (v - mu) * rs * g1[n] + beta1[n];
            sh[o] = gelu_exact(v);
        }
    }
    __syncthreads();
    // Linear(128,128) on matrix pipes: 8 N-tiles x 4 K-steps of v_wmma_f32_16x16x32_bf16
    v8f acc[8];
#pragma unroll
    for (int nt = 0; nt < 8; ++nt) {
        v8f c = {};
#pragma unroll
        for (int kt = 0; kt < 4; ++kt) {
            v16bf a  = load_a_frag(sh, stride, xoff, kt * 32, lane);
            v16bf bb = load_b_frag(w2p, kt, nt, 8, lane);
            c = __builtin_amdgcn_wmma_f32_16x16x32_bf16(false, a, false, bb, (short)0, c, false, false);
        }
        acc[nt] = c;
    }
    __syncthreads();
    // D layout: M = r + 8*(lane>=16), N = nt*16 + (lane&15); write x (+bias) back over h
#pragma unroll
    for (int nt = 0; nt < 8; ++nt) {
#pragma unroll
        for (int r = 0; r < 8; ++r) {
            int mm = r + (hf << 3);
            int n  = nt * 16 + m;
            sh[mm * stride + xoff + n] = acc[nt][r] + b2[n];
        }
    }
    __syncthreads();
}

// Kernel A: conv1 per tile, scatter-max x into centers y[B,1024,128]
__global__ void __launch_bounds__(32) kernelA(
    const float* __restrict__ pts, const long long* __restrict__ nn,
    const float* w1, const float* b1, const float* g1, const float* beta1,
    const __bf16* w2p, const float* b2, float* y)
{
    __shared__ float sh[16 * H0];
    __shared__ float stats[32];
    const int lane = threadIdx.x;
    const long long p0 = (long long)blockIdx.x * 16;
    compute_x_tile(sh, H0, 0, stats, p0, lane, pts, w1, b1, g1, beta1, w2p, b2);
    const int m  = lane & 15;
    const int hf = lane >> 4;
    const long long g = p0 + m;
    const int b  = (int)(g >> 16);         // 65536 points per batch
    const int ci = (int)nn[g];
    int* yrow = (int*)(y + (((long long)b * CNUM + ci) * H0));
    for (int j = 0; j < 64; ++j) {
        int n = hf * 64 + j;
        float v = sh[m * H0 + n];
        if (v > 0.f) atomicMax(&yrow[n], __float_as_int(v));  // monotone for positive floats
    }
}

// Kernel B: recompute x, gather x_max, conv2 MLP, scatter-max into output [B,1024,256]
__global__ void __launch_bounds__(32) kernelB(
    const float* __restrict__ pts, const long long* __restrict__ nn,
    const float* w1, const float* b1, const float* g1, const float* beta1,
    const __bf16* w2p, const float* b2,
    const __bf16* w3p, const float* b3, const float* g2, const float* beta2,
    const __bf16* w4p, const float* b4,
    const float* __restrict__ y, float* out)
{
    __shared__ float st[16 * H1];
    __shared__ float stats[32];
    __shared__ int   sidx[16];
    const int lane = threadIdx.x;
    const long long p0 = (long long)blockIdx.x * 16;
    const int b = (int)(p0 >> 16);
    if (lane < 16) sidx[lane] = (int)nn[p0 + lane];
    // x -> columns [128,256)
    compute_x_tile(st, H1, H0, stats, p0, lane, pts, w1, b1, g1, beta1, w2p, b2);
    const int m  = lane & 15;
    const int hf = lane >> 4;
    {   // gather x_max -> columns [0,128)
        const float* yrow = y + (((long long)b * CNUM + sidx[m]) * H0);
        for (int j = 0; j < 64; ++j) {
            int n = hf * 64 + j;
            st[m * H1 + n] = yrow[n];
        }
    }
    __syncthreads();
    // Linear(256,256): 16 N-tiles x 8 K-steps
    v8f acc[16];
#pragma unroll
    for (int nt = 0; nt < 16; ++nt) {
        v8f c = {};
#pragma unroll
        for (int kt = 0; kt < 8; ++kt) {
            __builtin_prefetch(w3p + ((((kt * 16 + nt) * 32) + lane) << 4) + 512, 0, 0);
            v16bf a  = load_a_frag(st, H1, 0, kt * 32, lane);
            v16bf bb = load_b_frag(w3p, kt, nt, 16, lane);
            c = __builtin_amdgcn_wmma_f32_16x16x32_bf16(false, a, false, bb, (short)0, c, false, false);
        }
        acc[nt] = c;
    }
    __syncthreads();
#pragma unroll
    for (int nt = 0; nt < 16; ++nt)
#pragma unroll
        for (int r = 0; r < 8; ++r) {
            int mm = r + (hf << 3);
            int n  = nt * 16 + m;
            st[mm * H1 + n] = acc[nt][r] + b3[n];
        }
    __syncthreads();
    // LayerNorm(256)
    if (lane < 16) {
        float s = 0.f, s2 = 0.f;
        for (int n = 0; n < H1; ++n) { float v = st[lane * H1 + n]; s += v; s2 += v * v; }
        float mu  = s * (1.0f / H1);
        float var = s2 * (1.0f / H1) - mu * mu;
        stats[lane]      = mu;
        stats[16 + lane] = rsqrtf(var + EPSL);
    }
    __syncthreads();
    {
        float mu = stats[m], rs = stats[16 + m];
        for (int j = 0; j < 128; ++j) {
            int n = hf * 128 + j;
            int o = m * H1 + n;
            float v = st[o];
            v = (v - mu) * rs * g2[n] + beta2[n];
            st[o] = gelu_exact(v);
        }
    }
    __syncthreads();
    // Linear(256,256) + fused scatter-max
#pragma unroll
    for (int nt = 0; nt < 16; ++nt) {
        v8f c = {};
#pragma unroll
        for (int kt = 0; kt < 8; ++kt) {
            v16bf a  = load_a_frag(st, H1, 0, kt * 32, lane);
            v16bf bb = load_b_frag(w4p, kt, nt, 16, lane);
            c = __builtin_amdgcn_wmma_f32_16x16x32_bf16(false, a, false, bb, (short)0, c, false, false);
        }
        acc[nt] = c;
    }
#pragma unroll
    for (int nt = 0; nt < 16; ++nt)
#pragma unroll
        for (int r = 0; r < 8; ++r) {
            int mm = r + (hf << 3);
            int n  = nt * 16 + m;
            float v = acc[nt][r] + b4[n];
            if (v > 0.f) {
                int* orow = (int*)(out + (((long long)b * CNUM + sidx[mm]) * H1));
                atomicMax(&orow[n], __float_as_int(v));
            }
        }
}

// Re-pack fp32 weights [K,N] into bf16 WMMA B-fragment order.
// element e: i=e&15, lane=(e>>4)&31, tile=e>>9; K = kt*32 + i + 16*(lane>=16), N = nt*16 + (lane&15)
__global__ void pack_w_bf16(const float* __restrict__ src, __bf16* __restrict__ dst, int K, int N) {
    int e = blockIdx.x * blockDim.x + threadIdx.x;
    int NT = N >> 4;
    int total = (K >> 5) * NT * 512;
    if (e >= total) return;
    int i    = e & 15;
    int lane = (e >> 4) & 31;
    int tile = e >> 9;
    int nt = tile % NT, kt = tile / NT;
    int k = kt * 32 + i + 16 * (lane >> 4);
    int n = nt * 16 + (lane & 15);
    dst[e] = (__bf16)src[k * N + n];
}

extern "C" void kernel_launch(void* const* d_in, const int* in_sizes, int n_in,
                              void* d_out, int out_size, void* d_ws, size_t ws_size,
                              hipStream_t stream) {
    (void)in_sizes; (void)n_in; (void)ws_size;
    const float*     pts = (const float*)d_in[0];
    const long long* nn  = (const long long*)d_in[1];   // jnp.int64
    const float* w1  = (const float*)d_in[3];
    const float* b1  = (const float*)d_in[4];
    const float* g1  = (const float*)d_in[5];
    const float* be1 = (const float*)d_in[6];
    const float* w2  = (const float*)d_in[7];
    const float* b2  = (const float*)d_in[8];
    const float* w3  = (const float*)d_in[9];
    const float* b3  = (const float*)d_in[10];
    const float* g2  = (const float*)d_in[11];
    const float* be2 = (const float*)d_in[12];
    const float* w4  = (const float*)d_in[13];
    const float* b4  = (const float*)d_in[14];
    float* out = (float*)d_out;

    // workspace layout: y[8*1024*128] f32, then packed bf16 weights
    char* ws = (char*)d_ws;
    const size_t yb = (size_t)8 * CNUM * H0 * sizeof(float);   // 4 MB
    float*  y   = (float*)ws;
    __bf16* w2p = (__bf16*)(ws + yb);                          // 16384  bf16
    __bf16* w3p = (__bf16*)(ws + yb + 32768);                  // 65536  bf16
    __bf16* w4p = (__bf16*)(ws + yb + 32768 + 131072);         // 65536  bf16

    hipMemsetAsync(y, 0, yb, stream);
    hipMemsetAsync(out, 0, (size_t)out_size * sizeof(float), stream);

    pack_w_bf16<<<(16384 + 255) / 256, 256, 0, stream>>>(w2, w2p, H0, H0);
    pack_w_bf16<<<(65536 + 255) / 256, 256, 0, stream>>>(w3, w3p, H1, H1);
    pack_w_bf16<<<(65536 + 255) / 256, 256, 0, stream>>>(w4, w4p, H1, H1);

    const int ntiles = (8 * 65536) / 16;  // 32768 point tiles
    kernelA<<<ntiles, 32, 0, stream>>>(pts, nn, w1, b1, g1, be1, w2p, b2, y);
    kernelB<<<ntiles, 32, 0, stream>>>(pts, nn, w1, b1, g1, be1, w2p, b2,
                                       w3p, b3, g2, be2, w4p, b4, y, out);
}
